// singleHeadAttention_11716670784129
// MI455X (gfx1250) — compile-verified, matrix operand order checked by reading, not run on hardware
//
#include <hip/hip_runtime.h>
#include <hip/hip_bf16.h>

// Sizes (fixed by the reference)
#define BB 8
#define TT 2048
#define CC 1024
#define HH 128

typedef __attribute__((ext_vector_type(16))) __bf16 v16bf;
typedef __attribute__((ext_vector_type(8)))  float  v8f;

// 16x32 bf16 A/B fragment: per-lane = one row (A: M=lane&15, B: N=lane&15),
// holding K = kk+0..7 (regs 0-3) and K = kk+16..23 (regs 4-7), kk = 8*(lane>>4).
union Frag16 { v16bf v; uint4 q[2]; unsigned u[8]; __bf16 h[16]; };

static __device__ __forceinline__ unsigned pack2bf(float a, float b) {
  unsigned short ua = __builtin_bit_cast(unsigned short, (__bf16)a);
  unsigned short ub = __builtin_bit_cast(unsigned short, (__bf16)b);
  return (unsigned)ua | ((unsigned)ub << 16);
}

// ---------------------------------------------------------------------------
// Kernel 1: kq = bf16(x @ Wk^T)  stored [B*T][H]   (q == k per reference bug)
//           vT = bf16(x @ Wv^T)  stored TRANSPOSED [B][H][T] so the attention
//           kernel's P@V B-fragments are contiguous global b128 loads.
// Block: 256 threads = 8 waves; block owns 16 M-rows; wave w owns N-cols 16w..16w+15.
// x tile staged to LDS via gfx1250 async-to-LDS DMA (no VGPR roundtrip).
// ---------------------------------------------------------------------------
__global__ __launch_bounds__(256)
void proj_kernel(const float* __restrict__ x, const float* __restrict__ Wk,
                 const float* __restrict__ Wv,
                 __bf16* __restrict__ kq, __bf16* __restrict__ vT) {
  __shared__ __align__(16) float xs[16 * 36];   // 16 rows x 32 fp32, stride 36 (144B: 16B-aligned chunks)

  const int tid  = threadIdx.x;
  const int wave = tid >> 5, lane = tid & 31;
  const int half = lane >> 4, n0 = lane & 15;
  const int kk   = half * 8;
  const int m0   = blockIdx.x * 16;
  const int hcol = wave * 16 + n0;
  const int bidx = m0 / TT, t0 = m0 % TT;

  // async staging assignment: 128 chunks of 16B cover the 16x32 fp32 tile
  const int srow = tid >> 3;          // valid when tid < 128
  const int scol = (tid & 7) * 4;

  v8f ck = {}; v8f cv = {};

  for (int k0 = 0; k0 < CC; k0 += 32) {
    __syncthreads();                  // previous iteration's LDS readers are done
    if (tid < 128) {
      const float* g = x + (size_t)(m0 + srow) * CC + k0 + scol;
      unsigned lds = (unsigned)(size_t)&xs[srow * 36 + scol];   // flat->LDS offset (low 32 bits)
      asm volatile("global_load_async_to_lds_b128 %0, %1, off"
                   :: "v"(lds), "v"((unsigned long long)(size_t)g) : "memory");
    }
    asm volatile("s_wait_asynccnt 0x0" ::: "memory");
    __syncthreads();                  // tile visible to all waves

    // A fragment: read fp32 from LDS, pack to bf16
    const float* xr = &xs[n0 * 36 + kk];
    const float4 f0 = *(const float4*)(xr);
    const float4 f1 = *(const float4*)(xr + 4);
    const float4 f2 = *(const float4*)(xr + 16);
    const float4 f3 = *(const float4*)(xr + 20);
    Frag16 a;
    a.u[0] = pack2bf(f0.x, f0.y); a.u[1] = pack2bf(f0.z, f0.w);
    a.u[2] = pack2bf(f1.x, f1.y); a.u[3] = pack2bf(f1.z, f1.w);
    a.u[4] = pack2bf(f2.x, f2.y); a.u[5] = pack2bf(f2.z, f2.w);
    a.u[6] = pack2bf(f3.x, f3.y); a.u[7] = pack2bf(f3.z, f3.w);

    // B fragments from Wk/Wv (fp32 global, L2-resident)
    Frag16 bk, bv;
    const float* wkp = Wk + (size_t)hcol * CC + k0 + kk;
    const float* wvp = Wv + (size_t)hcol * CC + k0 + kk;
#pragma unroll
    for (int i = 0; i < 8; ++i) { bk.h[i]     = (__bf16)wkp[i];      bv.h[i]     = (__bf16)wvp[i]; }
#pragma unroll
    for (int i = 0; i < 8; ++i) { bk.h[8 + i] = (__bf16)wkp[16 + i]; bv.h[8 + i] = (__bf16)wvp[16 + i]; }

    ck = __builtin_amdgcn_wmma_f32_16x16x32_bf16(false, a.v, false, bk.v, (short)0, ck, false, false);
    cv = __builtin_amdgcn_wmma_f32_16x16x32_bf16(false, a.v, false, bv.v, (short)0, cv, false, false);
  }

  // kq row-major [m][h]
#pragma unroll
  for (int r = 0; r < 8; ++r)
    kq[(size_t)(m0 + r + 8 * half) * HH + hcol] = (__bf16)ck[r];
  // vT transposed [b][h][t]: consecutive r are consecutive t -> packed b32 stores
  __bf16* vrow = vT + ((size_t)bidx * HH + hcol) * TT + t0 + 8 * half;
#pragma unroll
  for (int r = 0; r < 8; r += 2)
    *(unsigned*)&vrow[r] = pack2bf(cv[r], cv[r + 1]);
}

// ---------------------------------------------------------------------------
// Kernel 2: causal flash attention, 4 fully independent waves per block
// (16 q-rows each). K-tile = 32 keys: 8 WMMA for S, 8 WMMA for P@V per tile.
// No __syncthreads: V comes pre-transposed from global (L2-resident), only
// LDS use is the per-wave P layout bounce (C-layout -> A-layout).
// ---------------------------------------------------------------------------
__global__ __launch_bounds__(128)
void attn_kernel(const __bf16* __restrict__ kq, const __bf16* __restrict__ vT,
                 float* __restrict__ out) {
  __shared__ __align__(16) __bf16 pb[4 * 16 * 40];   // per-wave P bounce, stride 40

  const int tid  = threadIdx.x;
  const int wave = tid >> 5, lane = tid & 31;
  const int half = lane >> 4, n0 = lane & 15;
  const int kk   = half * 8;
  const int bb   = blockIdx.y;
  const int q0w  = blockIdx.x * 64 + wave * 16;
  const float scale = 0.03125f;                      // C^-0.5 = 1/32 (per reference)

  // Q A-fragments (q == k projection), resident for the whole kernel
  Frag16 aq[4];
  {
    const __bf16* qp = kq + ((size_t)bb * TT + q0w + n0) * HH;
#pragma unroll
    for (int hb = 0; hb < 4; ++hb) {
      aq[hb].q[0] = *(const uint4*)(qp + hb * 32 + kk);
      aq[hb].q[1] = *(const uint4*)(qp + hb * 32 + kk + 16);
    }
  }

  v8f o[8];
#pragma unroll
  for (int i = 0; i < 8; ++i) o[i] = (v8f){};
  float mrun[8], lrun[8];
#pragma unroll
  for (int r = 0; r < 8; ++r) { mrun[r] = -__builtin_inff(); lrun[r] = 0.f; }

  __bf16* pw = pb + wave * (16 * 40);
  const int nkt = q0w / 32 + 1;                      // per-wave causal tile count

  for (int j = 0; j < nkt; ++j) {
    const int k0 = j * 32;

    if (j + 1 < nkt) {                               // hint next tile into cache
      __builtin_prefetch(kq + ((size_t)bb * TT + k0 + 32 + lane) * HH, 0, 3);
      __builtin_prefetch(vT + ((size_t)bb * HH + lane * 4) * TT + k0 + 32, 0, 3);
    }

    // ---- S = q . k^T  (two 16x16 column groups) ----
    v8f s0 = {}; v8f s1 = {};
    const __bf16* kp = kq + ((size_t)bb * TT + k0) * HH;
#pragma unroll
    for (int hb = 0; hb < 4; ++hb) {
      Frag16 b0, b1;
      const __bf16* p0 = kp + (size_t)n0 * HH + hb * 32 + kk;
      const __bf16* p1 = kp + (size_t)(n0 + 16) * HH + hb * 32 + kk;
      b0.q[0] = *(const uint4*)(p0);      b0.q[1] = *(const uint4*)(p0 + 16);
      b1.q[0] = *(const uint4*)(p1);      b1.q[1] = *(const uint4*)(p1 + 16);
      s0 = __builtin_amdgcn_wmma_f32_16x16x32_bf16(false, aq[hb].v, false, b0.v, (short)0, s0, false, false);
      s1 = __builtin_amdgcn_wmma_f32_16x16x32_bf16(false, aq[hb].v, false, b1.v, (short)0, s1, false, false);
    }

    // ---- causal mask + online softmax; P -> per-wave LDS in A-layout source ----
    asm volatile("" ::: "memory");
#pragma unroll
    for (int r = 0; r < 8; ++r) {
      const int mg = q0w + r + 8 * half;             // global q row for this element
      float a = (k0 + n0      <= mg) ? s0[r] * scale : -__builtin_inff();
      float b = (k0 + 16 + n0 <= mg) ? s1[r] * scale : -__builtin_inff();
      float tm = fmaxf(a, b);
#pragma unroll
      for (int m = 1; m < 16; m <<= 1) tm = fmaxf(tm, __shfl_xor(tm, m, 32));
      const float mnew = fmaxf(mrun[r], tm);
      const float corr = __expf(mrun[r] - mnew);
      const float p0 = __expf(a - mnew);
      const float p1 = __expf(b - mnew);
      float rs = p0 + p1;
#pragma unroll
      for (int m = 1; m < 16; m <<= 1) rs += __shfl_xor(rs, m, 32);
      lrun[r] = lrun[r] * corr + rs;
      mrun[r] = mnew;
#pragma unroll
      for (int hb = 0; hb < 8; ++hb) o[hb][r] *= corr;
      const int ml = r + 8 * half;                   // local P row
      pw[ml * 40 + n0]      = (__bf16)p0;
      pw[ml * 40 + n0 + 16] = (__bf16)p1;
    }
    // same-wave DS ops are in order; wait for stores before re-reading
    asm volatile("s_wait_dscnt 0x0" ::: "memory");

    Frag16 pa;
    pa.q[0] = *(const uint4*)&pw[n0 * 40 + kk];
    pa.q[1] = *(const uint4*)&pw[n0 * 40 + kk + 16];

    // ---- O += P @ V : V B-fragments directly from transposed global vT ----
#pragma unroll
    for (int hb = 0; hb < 8; ++hb) {
      Frag16 vb;
      const __bf16* vp = vT + ((size_t)bb * HH + hb * 16 + n0) * TT + k0 + kk;
      vb.q[0] = *(const uint4*)(vp);
      vb.q[1] = *(const uint4*)(vp + 16);
      o[hb] = __builtin_amdgcn_wmma_f32_16x16x32_bf16(false, pa.v, false, vb.v, (short)0, o[hb], false, false);
    }
  }

  // ---- normalize and store fp32 ----
#pragma unroll
  for (int r = 0; r < 8; ++r) {
    const float inv = 1.0f / lrun[r];
    const size_t base = ((size_t)bb * TT + q0w + r + 8 * half) * HH;
#pragma unroll
    for (int hb = 0; hb < 8; ++hb)
      out[base + hb * 16 + n0] = o[hb][r] * inv;
  }
}

// ---------------------------------------------------------------------------
extern "C" void kernel_launch(void* const* d_in, const int* in_sizes, int n_in,
                              void* d_out, int out_size, void* d_ws, size_t ws_size,
                              hipStream_t stream) {
  (void)in_sizes; (void)n_in; (void)out_size; (void)ws_size;
  const float* x  = (const float*)d_in[0];
  const float* Wk = (const float*)d_in[1];
  // d_in[2] = Wq is intentionally unused (reference computes q with Wk)
  const float* Wv = (const float*)d_in[3];
  float* out = (float*)d_out;

  __bf16* kq = (__bf16*)d_ws;                          // [B,T,H] bf16 (4 MB)
  __bf16* vT = kq + (size_t)BB * TT * HH;              // [B,H,T] bf16 (4 MB, transposed)

  proj_kernel<<<dim3(BB * TT / 16), 256, 0, stream>>>(x, Wk, Wv, kq, vT);
  attn_kernel<<<dim3(TT / 64, BB), 128, 0, stream>>>(kq, vT, out);
}
